// LinearCrossEntropyLoss_17970143166524
// MI455X (gfx1250) — compile-verified, matrix operand order checked by reading, not run on hardware
//
#include <hip/hip_runtime.h>
#include <math.h>

typedef __attribute__((ext_vector_type(16))) __bf16 v16bf;
typedef __attribute__((ext_vector_type(8)))  float  v8f;
typedef __attribute__((ext_vector_type(4)))  unsigned int u32x4;
typedef __attribute__((ext_vector_type(4)))  int          i32x4;
typedef __attribute__((ext_vector_type(8)))  int          i32x8;

#define IGNORE_INDEX (-100)
#define D_DIM   2048
#define V_DIM   32000
#define ROWS_TOTAL 8192
#define TPB     256                    // 8 waves (wave32)
#define NWAVE   8
#define NTILE   16                     // vocab cols per WMMA tile
#define NT_TOTAL    (V_DIM / NTILE)    // 2000
#define NT_PER_WAVE (NT_TOTAL / NWAVE) // 250

// ---- fast path (pre-converted bf16 in d_ws) ----
#define MB_P    64                     // rows per workgroup
#define SMEM_P  (MB_P*D_DIM*2u + 2u*(NWAVE*MB_P*4u) + MB_P*4u + MB_P*4u)   // 266752 B

// ---- fallback path (convert W in-loop) ----
#define MB_F    32
#define SMEM_F  (MB_F*D_DIM*2u + 2u*(NWAVE*MB_F*4u) + MB_F*4u + MB_F*4u)   // 133376 B

// d_ws layout for fast path
#define W_BF16_BYTES  ((size_t)V_DIM * D_DIM * 2u)        // 131,072,000
#define A_BF16_BYTES  ((size_t)ROWS_TOTAL * D_DIM * 2u)   //  33,554,432
#define WS_NEEDED     (W_BF16_BYTES + A_BF16_BYTES + 16u)

#if defined(__gfx1250__) && __has_builtin(__builtin_amdgcn_tensor_load_to_lds) && \
    __has_builtin(__builtin_amdgcn_s_wait_tensorcnt)
#define USE_TDM 1
#else
#define USE_TDM 0
#endif

union BF16x16 {
    unsigned short u[16];
    v16bf          v;
    uint4          q[2];
};

__device__ __forceinline__ unsigned short bf16_rne(float f) {
    unsigned int u = __float_as_uint(f);
    return (unsigned short)((u + 0x7fffu + ((u >> 16) & 1u)) >> 16);
}

__device__ __forceinline__ BF16x16 ld_frag_lds(const unsigned short* p) {
    BF16x16 x;
    x.q[0] = ((const uint4*)p)[0];
    x.q[1] = ((const uint4*)p)[1];
    return x;
}

__device__ __forceinline__ BF16x16 ld_frag_glb(const unsigned short* p) {
    BF16x16 x;
    x.q[0] = ((const uint4*)p)[0];
    x.q[1] = ((const uint4*)p)[1];
    return x;
}

#if USE_TDM
// TDM: 1D tile copy of MB_P int32 targets (global -> LDS), D# per ISA 8.3/8.4.
__device__ __forceinline__ void tdm_load_targets(const int* targets_g, unsigned lds_off, int row0) {
    unsigned long long ga = (unsigned long long)(uintptr_t)(targets_g + row0);
    u32x4 g0;
    g0[0] = 1u;                                   // count=1, user mode
    g0[1] = lds_off;                              // lds_addr (bytes)
    g0[2] = (unsigned)ga;                         // global_addr[31:0]
    g0[3] = (unsigned)((ga >> 32) & 0x01ffffffu)  // global_addr[56:32]
          | (2u << 30);                           // type=2 ("image")
    i32x8 g1;
    g1[0] = (int)(2u << 16);                      // wg_mask=0, data_size=4B
    g1[1] = (int)((unsigned)(ROWS_TOTAL & 0xffff) << 16);      // tensor_dim0 lo
    g1[2] = (int)(((unsigned)ROWS_TOTAL >> 16) | (1u << 16));  // tensor_dim0 hi | tensor_dim1=1
    g1[3] = (int)((unsigned)MB_P << 16);          // tensor_dim1 hi=0 | tile_dim0=MB_P
    g1[4] = 1;                                    // tile_dim1=1, tile_dim2=0
    g1[5] = ROWS_TOTAL;                           // tensor_dim0_stride lo
    g1[6] = 0;
    g1[7] = 0;
    i32x4 z4 = {0, 0, 0, 0};
#if __has_include(<hip/amd_detail/amd_gfx1250_TDM.h>)
    i32x8 z8 = {0, 0, 0, 0, 0, 0, 0, 0};          // 6-arg toolchain (clang-23/therock)
    __builtin_amdgcn_tensor_load_to_lds(g0, g1, z4, z4, z8, 0);
#else
    __builtin_amdgcn_tensor_load_to_lds(g0, g1, z4, z4, 0);   // ROCm 7.2, 5-arg
#endif
    __builtin_amdgcn_s_wait_tensorcnt(0);
}
#endif

// ---------------------------------------------------------------------------
// Pre-pass: fp32 -> bf16 (RNE) for W and hidden, once per call.
// ---------------------------------------------------------------------------
extern "C" __global__ void lce_prep(const float* __restrict__ W,
                                    const float* __restrict__ hidden,
                                    unsigned short* __restrict__ Wb,
                                    unsigned short* __restrict__ Ab)
{
    const size_t stride = (size_t)gridDim.x * blockDim.x;
    size_t t = (size_t)blockIdx.x * blockDim.x + threadIdx.x;

    const size_t nw4 = (size_t)V_DIM * D_DIM / 4;
    const float4* w4 = (const float4*)W;
    for (size_t i = t; i < nw4; i += stride) {
        float4 x = w4[i];
        uint2 p;
        p.x = (unsigned)bf16_rne(x.x) | ((unsigned)bf16_rne(x.y) << 16);
        p.y = (unsigned)bf16_rne(x.z) | ((unsigned)bf16_rne(x.w) << 16);
        ((uint2*)Wb)[i] = p;
    }
    const size_t na4 = (size_t)ROWS_TOTAL * D_DIM / 4;
    const float4* h4 = (const float4*)hidden;
    for (size_t i = t; i < na4; i += stride) {
        float4 x = h4[i];
        uint2 p;
        p.x = (unsigned)bf16_rne(x.x) | ((unsigned)bf16_rne(x.y) << 16);
        p.y = (unsigned)bf16_rne(x.z) | ((unsigned)bf16_rne(x.w) << 16);
        ((uint2*)Ab)[i] = p;
    }
}

// ---------------------------------------------------------------------------
// Fast path: bf16 everywhere, zero conversion in the hot loop.
// 128 WGs x 64 rows; 8 waves each own a vocab slice; 4 row tiles per B fetch.
// ---------------------------------------------------------------------------
extern "C" __global__ void __launch_bounds__(TPB)
lce_main_bf16(const unsigned short* __restrict__ Ab,
              const int*            __restrict__ targets,
              const unsigned short* __restrict__ Wb,
              float*                __restrict__ accum)
{
    extern __shared__ char smem[];
    unsigned short* a_s = (unsigned short*)smem;              // MB_P*D
    float* stats_m = (float*)(a_s + MB_P * D_DIM);            // NWAVE*MB_P
    float* stats_s = stats_m + NWAVE * MB_P;
    float* tlogit  = stats_s + NWAVE * MB_P;                  // MB_P
    int*   tgt     = (int*)(tlogit + MB_P);                   // MB_P

    const int tid  = threadIdx.x;
    const int lane = tid & 31;
    const int wid  = tid >> 5;
    const int l16  = lane & 15;
    const int hseg = lane >> 4;
    const int row0 = blockIdx.x * MB_P;

#if USE_TDM
    if (wid == 0) tdm_load_targets(targets, (unsigned)(uintptr_t)tgt, row0);
    if (tid < MB_P) tlogit[tid] = 0.0f;
#else
    if (tid < MB_P) { tgt[tid] = targets[row0 + tid]; tlogit[tid] = 0.0f; }
#endif

    // Stage pre-converted bf16 hidden block into LDS with bank swizzle.
    {
        const uint2* g2 = (const uint2*)(Ab + (size_t)row0 * D_DIM);
        for (int i = tid; i < (MB_P * D_DIM) / 4; i += TPB) {
            uint2 x = g2[i];
            int idx   = i * 4;
            int row   = idx >> 11;               // / D_DIM
            int k     = idx & (D_DIM - 1);
            int chunk = (k >> 4) ^ (row & 7);    // 16-elem chunk swizzle
            int pos   = row * D_DIM + chunk * 16 + (k & 15);
            *(uint2*)(a_s + pos) = x;
        }
    }
    __syncthreads();

    // Per-lane online softmax state: 4 row tiles x 8 accumulator rows = 32 slots.
    // slot = t*8 + v  ->  local row = 16*t + v + 8*hseg ; this lane's col only.
    float m_run[32], s_run[32];
    #pragma unroll
    for (int j = 0; j < 32; ++j) { m_run[j] = -3.0e38f; s_run[j] = 0.0f; }

    for (int it = 0; it < NT_PER_WAVE; ++it) {
        const int n0  = (wid + NWAVE * it) * NTILE;
        const int col = n0 + l16;
        const unsigned short* wrow = Wb + (size_t)col * D_DIM + (hseg << 4);

        v8f c0 = {0,0,0,0,0,0,0,0};
        v8f c1 = {0,0,0,0,0,0,0,0};
        v8f c2 = {0,0,0,0,0,0,0,0};
        v8f c3 = {0,0,0,0,0,0,0,0};

        for (int k0 = 0; k0 < D_DIM; k0 += 32) {
            BF16x16 b = ld_frag_glb(wrow + k0);           // 2x b128, bf16 direct
            if (k0 + 512 < D_DIM) __builtin_prefetch(wrow + k0 + 512, 0, 2);

            // All four rows are congruent mod 8 -> identical swizzle term;
            // batch all fragment loads before the WMMAs so the dscnt waits
            // overlap LDS latency with XDL execution.
            const int kc = (k0 >> 4) + hseg;
            const unsigned short* pa = a_s + l16 * D_DIM + ((kc ^ (l16 & 7)) << 4);
            BF16x16 a0 = ld_frag_lds(pa);
            BF16x16 a1 = ld_frag_lds(pa + 16 * D_DIM);
            BF16x16 a2 = ld_frag_lds(pa + 32 * D_DIM);
            BF16x16 a3 = ld_frag_lds(pa + 48 * D_DIM);

            c0 = __builtin_amdgcn_wmma_f32_16x16x32_bf16(false, a0.v, false, b.v, (short)0, c0, false, false);
            c1 = __builtin_amdgcn_wmma_f32_16x16x32_bf16(false, a1.v, false, b.v, (short)0, c1, false, false);
            c2 = __builtin_amdgcn_wmma_f32_16x16x32_bf16(false, a2.v, false, b.v, (short)0, c2, false, false);
            c3 = __builtin_amdgcn_wmma_f32_16x16x32_bf16(false, a3.v, false, b.v, (short)0, c3, false, false);
        }

        // Per-lane online-softmax update (no shuffles / no cross-lane traffic).
        #pragma unroll
        for (int v = 0; v < 8; ++v) {
            float xs[4] = {c0[v], c1[v], c2[v], c3[v]};
            #pragma unroll
            for (int t = 0; t < 4; ++t) {
                const int slot = t * 8 + v;
                const int rl   = 16 * t + v + (hseg << 3);
                float x  = xs[t];
                float nm = fmaxf(m_run[slot], x);
                s_run[slot] = s_run[slot] * __expf(m_run[slot] - nm) + __expf(x - nm);
                m_run[slot] = nm;
                if (tgt[rl] == col) tlogit[rl] = x;        // unique writer
            }
        }
    }

    // One-time cross-lane merge of (m, s) within each 16-lane half.
    #pragma unroll
    for (int j = 0; j < 32; ++j) {
        float m = m_run[j], s = s_run[j];
        #pragma unroll
        for (int d = 1; d < 16; d <<= 1) {
            float mo = __shfl_xor(m, d, 16);
            float so = __shfl_xor(s, d, 16);
            float M  = fmaxf(m, mo);
            s = s * __expf(m - M) + so * __expf(mo - M);
            m = M;
        }
        m_run[j] = m; s_run[j] = s;
    }

    if (l16 == 0) {
        #pragma unroll
        for (int j = 0; j < 32; ++j) {
            int rl = 16 * (j >> 3) + (j & 7) + (hseg << 3);
            stats_m[wid * MB_P + rl] = m_run[j];
            stats_s[wid * MB_P + rl] = s_run[j];
        }
    }
    __syncthreads();

    if (tid < MB_P) {
        float M = -3.0e38f;
        #pragma unroll
        for (int w = 0; w < NWAVE; ++w) M = fmaxf(M, stats_m[w * MB_P + tid]);
        float S = 0.0f;
        #pragma unroll
        for (int w = 0; w < NWAVE; ++w) S += stats_s[w * MB_P + tid] * __expf(stats_m[w * MB_P + tid] - M);
        if (tgt[tid] != IGNORE_INDEX) {
            float loss = __logf(S) + M - tlogit[tid];
            atomicAdd(&accum[0], loss);
            atomicAdd(&accum[1], 1.0f);
        }
    }
}

// ---------------------------------------------------------------------------
// Fallback (ws too small): convert W hi-bf16 in-loop, 32 rows/WG, 2 row tiles.
// ---------------------------------------------------------------------------
extern "C" __global__ void __launch_bounds__(TPB)
lce_main_fb(const float* __restrict__ hidden,
            const int*   __restrict__ targets,
            const float* __restrict__ W,
            float*       __restrict__ accum)
{
    extern __shared__ char smem[];
    unsigned short* a_s = (unsigned short*)smem;              // MB_F*D
    float* stats_m = (float*)(a_s + MB_F * D_DIM);
    float* stats_s = stats_m + NWAVE * MB_F;
    float* tlogit  = stats_s + NWAVE * MB_F;
    int*   tgt     = (int*)(tlogit + MB_F);

    const int tid  = threadIdx.x;
    const int lane = tid & 31;
    const int wid  = tid >> 5;
    const int l16  = lane & 15;
    const int hseg = lane >> 4;
    const int row0 = blockIdx.x * MB_F;

    if (tid < MB_F) { tgt[tid] = targets[row0 + tid]; tlogit[tid] = 0.0f; }

    const float4* h4 = (const float4*)(hidden + (size_t)row0 * D_DIM);
    for (int i = tid; i < (MB_F * D_DIM) / 4; i += TPB) {
        float4 x = h4[i];
        int idx   = i * 4;
        int row   = idx >> 11;
        int k     = idx & (D_DIM - 1);
        int chunk = (k >> 4) ^ (row & 7);
        int pos   = row * D_DIM + chunk * 16 + (k & 15);
        uint2 p;
        p.x = (unsigned)bf16_rne(x.x) | ((unsigned)bf16_rne(x.y) << 16);
        p.y = (unsigned)bf16_rne(x.z) | ((unsigned)bf16_rne(x.w) << 16);
        *(uint2*)(a_s + pos) = p;
    }
    __syncthreads();

    float m_run[16], s_run[16];
    #pragma unroll
    for (int j = 0; j < 16; ++j) { m_run[j] = -3.0e38f; s_run[j] = 0.0f; }

    for (int it = 0; it < NT_PER_WAVE; ++it) {
        const int n0  = (wid + NWAVE * it) * NTILE;
        const int col = n0 + l16;
        const float* wrow = W + (size_t)col * D_DIM + (hseg << 4);

        v8f c0 = {0,0,0,0,0,0,0,0};
        v8f c1 = {0,0,0,0,0,0,0,0};

        for (int k0 = 0; k0 < D_DIM; k0 += 32) {
            const float4* wp = (const float4*)(wrow + k0);
            float4 w0 = wp[0], w1 = wp[1], w2 = wp[2], w3 = wp[3];
            if (k0 + 512 < D_DIM) __builtin_prefetch(wrow + k0 + 512, 0, 2);

            BF16x16 b;
            {
                float wf[16] = {w0.x,w0.y,w0.z,w0.w, w1.x,w1.y,w1.z,w1.w,
                                w2.x,w2.y,w2.z,w2.w, w3.x,w3.y,w3.z,w3.w};
                #pragma unroll
                for (int j = 0; j < 16; ++j) b.u[j] = bf16_rne(wf[j]);
            }

            const int kc = (k0 >> 4) + hseg;
            const int r0 = l16, r1 = l16 + 16;
            BF16x16 a0 = ld_frag_lds(a_s + r0 * D_DIM + ((kc ^ (r0 & 7)) << 4));
            BF16x16 a1 = ld_frag_lds(a_s + r1 * D_DIM + ((kc ^ (r1 & 7)) << 4));

            c0 = __builtin_amdgcn_wmma_f32_16x16x32_bf16(false, a0.v, false, b.v, (short)0, c0, false, false);
            c1 = __builtin_amdgcn_wmma_f32_16x16x32_bf16(false, a1.v, false, b.v, (short)0, c1, false, false);
        }

        #pragma unroll
        for (int v = 0; v < 8; ++v) {
            {
                float x  = c0[v];
                float nm = fmaxf(m_run[v], x);
                s_run[v] = s_run[v] * __expf(m_run[v] - nm) + __expf(x - nm);
                m_run[v] = nm;
                int rl = v + (hseg << 3);
                if (tgt[rl] == col) tlogit[rl] = x;
            }
            {
                float x  = c1[v];
                float nm = fmaxf(m_run[v + 8], x);
                s_run[v + 8] = s_run[v + 8] * __expf(m_run[v + 8] - nm) + __expf(x - nm);
                m_run[v + 8] = nm;
                int rl = 16 + v + (hseg << 3);
                if (tgt[rl] == col) tlogit[rl] = x;
            }
        }
    }

    #pragma unroll
    for (int j = 0; j < 16; ++j) {
        float m = m_run[j], s = s_run[j];
        #pragma unroll
        for (int d = 1; d < 16; d <<= 1) {
            float mo = __shfl_xor(m, d, 16);
            float so = __shfl_xor(s, d, 16);
            float M  = fmaxf(m, mo);
            s = s * __expf(m - M) + so * __expf(mo - M);
            m = M;
        }
        m_run[j] = m; s_run[j] = s;
    }

    if (l16 == 0) {
        #pragma unroll
        for (int j = 0; j < 16; ++j) {
            int rl = 16 * (j >> 3) + (j & 7) + (hseg << 3);
            stats_m[wid * MB_F + rl] = m_run[j];
            stats_s[wid * MB_F + rl] = s_run[j];
        }
    }
    __syncthreads();

    if (tid < MB_F) {
        float M = -3.0e38f;
        #pragma unroll
        for (int w = 0; w < NWAVE; ++w) M = fmaxf(M, stats_m[w * MB_F + tid]);
        float S = 0.0f;
        #pragma unroll
        for (int w = 0; w < NWAVE; ++w) S += stats_s[w * MB_F + tid] * __expf(stats_m[w * MB_F + tid] - M);
        if (tgt[tid] != IGNORE_INDEX) {
            float loss = __logf(S) + M - tlogit[tid];
            atomicAdd(&accum[0], loss);
            atomicAdd(&accum[1], 1.0f);
        }
    }
}

extern "C" __global__ void lce_init(float* accum) {
    accum[0] = 0.0f; accum[1] = 0.0f;
}

extern "C" __global__ void lce_fin(const float* __restrict__ accum, float* __restrict__ out) {
    out[0] = accum[0] / accum[1];
}

extern "C" void kernel_launch(void* const* d_in, const int* in_sizes, int n_in,
                              void* d_out, int out_size, void* d_ws, size_t ws_size,
                              hipStream_t stream) {
    const float* hidden  = (const float*)d_in[0];
    const int*   targets = (const int*)  d_in[1];
    const float* W       = (const float*)d_in[2];
    float* out = (float*)d_out;

    (void)hipFuncSetAttribute((const void*)lce_main_bf16,
                              hipFuncAttributeMaxDynamicSharedMemorySize, (int)SMEM_P);
    (void)hipFuncSetAttribute((const void*)lce_main_fb,
                              hipFuncAttributeMaxDynamicSharedMemorySize, (int)SMEM_F);

    if (ws_size >= WS_NEEDED) {
        // Fast path: one-time bf16 conversion into scratch, conversion-free hot loop.
        unsigned short* Wb = (unsigned short*)d_ws;
        unsigned short* Ab = (unsigned short*)((char*)d_ws + W_BF16_BYTES);
        float* accum = (float*)((char*)d_ws + W_BF16_BYTES + A_BF16_BYTES);

        lce_prep<<<4096, TPB, 0, stream>>>(W, hidden, Wb, Ab);
        lce_init<<<1, 1, 0, stream>>>(accum);
        lce_main_bf16<<<ROWS_TOTAL / MB_P, TPB, SMEM_P, stream>>>(Ab, targets, Wb, accum);
        lce_fin<<<1, 1, 0, stream>>>(accum, out);
    } else {
        float* accum = (float*)d_ws;   // needs only 8 bytes
        lce_init<<<1, 1, 0, stream>>>(accum);
        lce_main_fb<<<ROWS_TOTAL / MB_F, TPB, SMEM_F, stream>>>(hidden, targets, W, accum);
        lce_fin<<<1, 1, 0, stream>>>(accum, out);
    }
}
